// Attention_model_44057774522458
// MI455X (gfx1250) — matrix operand order repaired and checked by reading, not compiled, not run on hardware
//
#include <hip/hip_runtime.h>
#include <hip/hip_bf16.h>

typedef _Float16 half8 __attribute__((ext_vector_type(8)));
typedef _Float16 v16h  __attribute__((ext_vector_type(16)));
typedef float    v8f   __attribute__((ext_vector_type(8)));

#define B_ 512
#define L_ 64
#define D_ 1024
#define U_ 512

// ---------------------------------------------------------------------------
// Prep kernels: f32 -> f16 conversion, and transpose+convert for the weight
// matrices so WMMA B-fragments are K-contiguous per output column.
// ---------------------------------------------------------------------------
__global__ void cvt_f16_kernel(const float* __restrict__ in,
                               _Float16* __restrict__ out, int n) {
    int i = blockIdx.x * blockDim.x + threadIdx.x;
    int stride = gridDim.x * blockDim.x;
    for (; i < n; i += stride) out[i] = (_Float16)in[i];
}

// out[n*rows + k] = (f16) in[k*cols + n]   (in: [rows=K, cols=N] row-major)
__global__ void transpose_f16_kernel(const float* __restrict__ in,
                                     _Float16* __restrict__ out,
                                     int rows, int cols) {
    int i = blockIdx.x * blockDim.x + threadIdx.x;
    int total = rows * cols;
    int stride = gridDim.x * blockDim.x;
    for (; i < total; i += stride) {
        int n = i / rows;
        int k = i - n * rows;
        out[i] = (_Float16)in[(size_t)k * cols + n];
    }
}

// ---------------------------------------------------------------------------
// WMMA fragment loaders (wave32, v_wmma_f32_16x16x32_f16 layouts).
// A (16x32, row-major source): lane = 16*g + m ;
//   elements j<8  -> K = k0 + 8*g + j ; j>=8 -> K = k0 + 16 + 8*g + (j-8)
// B (32x16, from K-contiguous transposed source): lane = 16*g + n ;
//   elements j<8  -> K = k0 + 8*g + j ; j>=8 -> K = k0 + 16 + 8*g + (j-8)
// ---------------------------------------------------------------------------
__device__ __forceinline__ v16h load_a_frag(const _Float16* __restrict__ row,
                                            int k0, int g) {
    v16h a;
    *(half8*)&a        = *(const half8*)(row + k0 + 8 * g);
    *((half8*)&a + 1)  = *(const half8*)(row + k0 + 16 + 8 * g);
    return a;
}

__device__ __forceinline__ v16h load_b_frag(const _Float16* __restrict__ col,
                                            int k0, int g) {
    v16h b;
    *(half8*)&b        = *(const half8*)(col + k0 + 8 * g);
    *((half8*)&b + 1)  = *(const half8*)(col + k0 + 16 + 8 * g);
    return b;
}

// ---------------------------------------------------------------------------
// hpre = hidden @ W2 + b2     [B,U] = [B,U]x[U,U]
// Grid: B/64 workgroups of 256 threads; wave w owns columns [w*64, w*64+64).
// ---------------------------------------------------------------------------
__global__ __launch_bounds__(256) void hidden_gemm_kernel(
    const _Float16* __restrict__ Hh,    // [B,U] f16
    const _Float16* __restrict__ W2T,   // [U,U] f16, n-major (K contiguous)
    const float* __restrict__ W2b,      // [U]
    float* __restrict__ hpre)           // [B,U] f32
{
    const int tid  = threadIdx.x;
    const int wave = tid >> 5;
    const int lane = tid & 31;
    const int g    = lane >> 4;
    const int ln   = lane & 15;
    const int m_base = blockIdx.x * 64;
    const int n0     = wave * 64;

    v8f acc[4][4] = {};

    for (int k0 = 0; k0 < U_; k0 += 32) {
        v16h afrag[4];
#pragma unroll
        for (int mt = 0; mt < 4; ++mt)
            afrag[mt] = load_a_frag(Hh + (size_t)(m_base + mt * 16 + ln) * U_, k0, g);
#pragma unroll
        for (int nt = 0; nt < 4; ++nt) {
            const int n = n0 + nt * 16 + ln;
            v16h bfrag = load_b_frag(W2T + (size_t)n * U_, k0, g);
#pragma unroll
            for (int mt = 0; mt < 4; ++mt)
                acc[mt][nt] = __builtin_amdgcn_wmma_f32_16x16x32_f16(
                    false, afrag[mt], false, bfrag, (short)0, acc[mt][nt],
                    false, false);
        }
    }

    // C/D layout: VGPR r of lane (16*g + ln) holds (M = mt*16 + 8*g + r, N = n0 + nt*16 + ln)
#pragma unroll
    for (int nt = 0; nt < 4; ++nt) {
        const int n = n0 + nt * 16 + ln;
        const float bias = W2b[n];
#pragma unroll
        for (int mt = 0; mt < 4; ++mt) {
#pragma unroll
            for (int r = 0; r < 8; ++r) {
                const int m = m_base + mt * 16 + 8 * g + r;
                hpre[(size_t)m * U_ + n] = acc[mt][nt][r] + bias;
            }
        }
    }
}

// ---------------------------------------------------------------------------
// Fused attention kernel: one workgroup per batch row b.
//   score  = tanh(F_b @ W1 + b1 + h_b)       (never materialized)
//   logits = score @ V                        (LDS ds_add_f32 reduction)
//   attn   = softmax(logits)                  (over L=64)
//   ctx    = attn^T @ F_b (fp32 features)
// ---------------------------------------------------------------------------
__global__ __launch_bounds__(256) void attn_fused_kernel(
    const _Float16* __restrict__ Fh,    // [B,L,D] f16
    const float*    __restrict__ F32,   // [B,L,D] f32 (original)
    const _Float16* __restrict__ W1T,   // [U,D] f16, n-major (K contiguous)
    const float* __restrict__ W1b,      // [U]
    const float* __restrict__ hpre,     // [B,U]
    const float* __restrict__ Vw,       // [U]
    float* __restrict__ outCtx,         // [B,D]
    float* __restrict__ outAttn)        // [B,L]
{
    __shared__ float s_logits[L_];
    __shared__ float s_attn[L_];

    const int b    = blockIdx.x;
    const int tid  = threadIdx.x;
    const int wave = tid >> 5;
    const int lane = tid & 31;
    const int g    = lane >> 4;
    const int ln   = lane & 15;
    const int n0   = wave * 64;         // this wave's 64-column slab of U=512

    if (tid < L_) s_logits[tid] = 0.0f;

    const _Float16* __restrict__ Fb = Fh + (size_t)b * L_ * D_;

    v8f acc[4][4] = {};

    for (int k0 = 0; k0 < D_; k0 += 32) {
        v16h afrag[4];
#pragma unroll
        for (int mt = 0; mt < 4; ++mt)
            afrag[mt] = load_a_frag(Fb + (size_t)(mt * 16 + ln) * D_, k0, g);
#pragma unroll
        for (int nt = 0; nt < 4; ++nt) {
            const int n = n0 + nt * 16 + ln;
            v16h bfrag = load_b_frag(W1T + (size_t)n * D_, k0, g);
#pragma unroll
            for (int mt = 0; mt < 4; ++mt)
                acc[mt][nt] = __builtin_amdgcn_wmma_f32_16x16x32_f16(
                    false, afrag[mt], false, bfrag, (short)0, acc[mt][nt],
                    false, false);
        }
    }

    // Per-lane epilogue: tanh + dot with V, reduce into LDS logits.
    float bias[4], vs[4];
#pragma unroll
    for (int nt = 0; nt < 4; ++nt) {
        const int n = n0 + nt * 16 + ln;
        bias[nt] = W1b[n] + hpre[(size_t)b * U_ + n];
        vs[nt]   = Vw[n];
    }

    __syncthreads();   // logits init done (pre-GEMM in program order)

#pragma unroll
    for (int mt = 0; mt < 4; ++mt) {
#pragma unroll
        for (int r = 0; r < 8; ++r) {
            const int m = mt * 16 + 8 * g + r;
            float s = 0.0f;
#pragma unroll
            for (int nt = 0; nt < 4; ++nt)
                s += tanhf(acc[mt][nt][r] + bias[nt]) * vs[nt];
            atomicAdd(&s_logits[m], s);   // ds_add_f32
        }
    }
    __syncthreads();

    // Softmax over L=64 (redundant per-thread scan; V_b cancels in softmax).
    float mx = -3.0e38f;
    for (int i = 0; i < L_; ++i) mx = fmaxf(mx, s_logits[i]);
    float sum = 0.0f;
    for (int i = 0; i < L_; ++i) sum += __expf(s_logits[i] - mx);
    const float inv = 1.0f / sum;

    if (tid < L_) {
        const float a = __expf(s_logits[tid] - mx) * inv;
        s_attn[tid] = a;
        outAttn[(size_t)b * L_ + tid] = a;
    }
    __syncthreads();

    // context[b][d] = sum_m attn[m] * F32[b][m][d]  (full-precision features)
    const float* __restrict__ F32b = F32 + (size_t)b * L_ * D_;
    for (int d = tid; d < D_; d += 256) {
        float c = 0.0f;
#pragma unroll 8
        for (int m = 0; m < L_; ++m)
            c += s_attn[m] * F32b[(size_t)m * D_ + d];
        outCtx[(size_t)b * D_ + d] = c;
    }
}

// ---------------------------------------------------------------------------
extern "C" void kernel_launch(void* const* d_in, const int* in_sizes, int n_in,
                              void* d_out, int out_size, void* d_ws, size_t ws_size,
                              hipStream_t stream) {
    const float* features = (const float*)d_in[0];   // [B,L,D]
    const float* hidden   = (const float*)d_in[1];   // [B,U]
    const float* W1w      = (const float*)d_in[2];   // [D,U]
    const float* W1b      = (const float*)d_in[3];   // [U]
    const float* W2w      = (const float*)d_in[4];   // [U,U]
    const float* W2b      = (const float*)d_in[5];   // [U]
    const float* Vw       = (const float*)d_in[6];   // [U,1]
    // d_in[7] = V_b: softmax(logits + c) == softmax(logits); not needed.

    float* outCtx  = (float*)d_out;                       // [B,D]
    float* outAttn = (float*)d_out + (size_t)B_ * D_;     // [B,L]

    // Workspace partition (~70.2 MB total)
    char* ws = (char*)d_ws;
    _Float16* Fh  = (_Float16*)ws;  ws += (size_t)B_ * L_ * D_ * sizeof(_Float16);
    _Float16* W1T = (_Float16*)ws;  ws += (size_t)D_ * U_ * sizeof(_Float16);
    _Float16* Hh  = (_Float16*)ws;  ws += (size_t)B_ * U_ * sizeof(_Float16);
    _Float16* W2T = (_Float16*)ws;  ws += (size_t)U_ * U_ * sizeof(_Float16);
    float*    hpre = (float*)ws;    ws += (size_t)B_ * U_ * sizeof(float);

    // Prep: conversions + weight transposes
    cvt_f16_kernel<<<8192, 256, 0, stream>>>(features, Fh, B_ * L_ * D_);
    cvt_f16_kernel<<<1024, 256, 0, stream>>>(hidden, Hh, B_ * U_);
    transpose_f16_kernel<<<2048, 256, 0, stream>>>(W1w, W1T, D_, U_);
    transpose_f16_kernel<<<1024, 256, 0, stream>>>(W2w, W2T, U_, U_);

    // h = hidden @ W2 + b2
    hidden_gemm_kernel<<<B_ / 64, 256, 0, stream>>>(Hh, W2T, W2b, hpre);

    // Fused attention: one workgroup per batch row
    attn_fused_kernel<<<B_, 256, 0, stream>>>(Fh, features, W1T, W1b, hpre,
                                              Vw, outCtx, outAttn);
}
